// SWA_927712935974
// MI455X (gfx1250) — matrix operand order repaired and checked by reading, not compiled, not run on hardware
//
#include <hip/hip_runtime.h>
#include <hip/hip_bf16.h>
#include <cstdint>
#include <cstddef>

// ---------------------------------------------------------------------------
// Sliding-window GQA attention for MI455X (gfx1250, wave32, WMMA).
// Pipeline: f32->f16 cast, WMMA GEMM qkv projections (128x256 per block,
// 16 WMMAs/wave/k-step, block-shared B slab), flash-attention with online
// softmax (WMMA for QK^T and PV), WMMA GEMM output projection (f32 out).
// ---------------------------------------------------------------------------

typedef __attribute__((ext_vector_type(16))) _Float16 v16h;
typedef __attribute__((ext_vector_type(8)))  _Float16 v8h;
typedef __attribute__((ext_vector_type(8)))  float    v8f;

constexpr int Bn  = 2;
constexpr int Ln  = 2048;
constexpr int En  = 2048;
constexpr int Hn  = 16;
constexpr int KVH = 4;
constexpr int HD  = 128;
constexpr int WIN = 512;
constexpr float SCALE  = 0.08838834764831845f;   // 1/sqrt(128)
constexpr float NEGBIG = -1e30f;

#define DEVINL __device__ __forceinline__

DEVINL v8f wmma_f16(v16h a, v16h b, v8f c) {
  // v_wmma_f32_16x16x32_f16 : D = A(16x32) * B(32x16) + C(16x16)
  return __builtin_amdgcn_wmma_f32_16x16x32_f16(
      /*neg_a=*/false, a, /*neg_b=*/false, b,
      /*c_mod=*/(short)0, c, /*reuse_a=*/false, /*reuse_b=*/false);
}

DEVINL void wait_ds() { asm volatile("s_wait_dscnt 0x0" ::: "memory"); }

// ---------------------------------------------------------------------------
// fp32 -> fp16 cast (grid-stride)
// ---------------------------------------------------------------------------
__global__ void swa_cvt_f32_to_f16(const float* __restrict__ src,
                                   _Float16* __restrict__ dst, int n) {
  int i = blockIdx.x * blockDim.x + threadIdx.x;
  int stride = gridDim.x * blockDim.x;
  for (; i < n; i += stride) dst[i] = (_Float16)src[i];
}

// ---------------------------------------------------------------------------
// Tiled WMMA GEMM: C[M,N] = A[M,K] * B[K,N], f16 inputs.
// Block = 8 waves (256 threads) computing a 128x256 C tile:
//   - wave w owns rows [m0b+16w, +16) x all 256 cols: 16 f32 accumulators.
//   - per k-step (32): each wave stages its own 16x32 A tile; the 32x256 B
//     slab is staged cooperatively (transposed to [n][k]) and shared by all
//     waves via the workgroup split-barrier, so every WMMA B-fragment read is
//     one contiguous pair of ds_load_b128.
//   - each wave then issues 16 WMMAs off a single A fragment.
// Arithmetic intensity ~89 flop/byte of L2 traffic (vs 15 for 16x256 strips),
// which moves the GEMMs from L2-bound to near the WMMA roofline.
// ---------------------------------------------------------------------------
template <bool F32OUT>
__global__ __launch_bounds__(256) void swa_gemm_wmma(
    const _Float16* __restrict__ A, const _Float16* __restrict__ Bm,
    void* __restrict__ Cout, int M, int N, int K) {
  __shared__ __align__(32) _Float16 sA[8][16 * 32];   // per-wave A tiles
  __shared__ __align__(32) _Float16 sBT[256 * 32];    // [n][k], block-shared

  const int tid  = threadIdx.x;
  const int wave = tid >> 5;
  const int lane = tid & 31;
  const int nblk = N >> 8;                      // # of 256-col block strips
  const int m0b = (blockIdx.x / nblk) << 7;     // 128-row block tile
  const int n0b = (blockIdx.x % nblk) << 8;     // 256-col block tile
  const int m0w = m0b + wave * 16;              // this wave's 16 rows
  const int row = lane & 15;
  const int grp = lane >> 4;   // lane group (0: lanes 0-15, 1: lanes 16-31)

  // B staging map: thread t loads B row (t>>3), col chunk (t&7)*32 (32 halves)
  const int bk  = tid >> 3;         // 0..31 : k row within slab
  const int bc0 = (tid & 7) * 32;   // 0..224: first col of this thread's chunk

  v8f acc[16];
#pragma unroll
  for (int nt = 0; nt < 16; ++nt) acc[nt] = (v8f){};

  for (int k0 = 0; k0 < K; k0 += 32) {
    // ---- stage this wave's A tile 16x32 ([row][k]) ----
    v16h av = *(const v16h*)(A + (size_t)(m0w + row) * K + k0 + grp * 16);
    *(v16h*)&sA[wave][row * 32 + grp * 16] = av;

    // ---- stage B slab 32x256 transposed to [n][k] (block-cooperative) ----
    const _Float16* brow = Bm + (size_t)(k0 + bk) * N + n0b + bc0;
    v16h bv0 = *(const v16h*)(brow);
    v16h bv1 = *(const v16h*)(brow + 16);
#pragma unroll
    for (int e = 0; e < 16; ++e) sBT[(bc0 + e) * 32 + bk] = bv0[e];
#pragma unroll
    for (int e = 0; e < 16; ++e) sBT[(bc0 + 16 + e) * 32 + bk] = bv1[e];

    if (k0 + 32 < K) {
      __builtin_prefetch(A + (size_t)(m0w + row) * K + k0 + 32 + grp * 16, 0, 0);
      __builtin_prefetch(brow + (size_t)32 * N, 0, 0);
    }
    wait_ds();
    __syncthreads();   // A tiles + shared B slab visible to all waves

    // ---- A fragment: lanes<16 K=[0..15], lanes>=16 K=[16..31]
    v8h alo = *(const v8h*)&sA[wave][row * 32 + grp * 16];
    v8h ahi = *(const v8h*)&sA[wave][row * 32 + grp * 16 + 8];
    v16h afrag = __builtin_shufflevector(alo, ahi, 0, 1, 2, 3, 4, 5, 6, 7, 8, 9,
                                         10, 11, 12, 13, 14, 15);
    // ---- 16 WMMAs off one A fragment; B frag = contiguous b128 LDS read ----
#pragma unroll
    for (int nt = 0; nt < 16; ++nt) {
      v16h bfrag = *(const v16h*)&sBT[(nt * 16 + row) * 32 + grp * 16];
      acc[nt] = wmma_f16(afrag, bfrag, acc[nt]);
    }
    __syncthreads();   // all reads done before next k-step restages
  }

  // ---- C epilogue: VGPR r -> M=r+8*grp, N = lane&15 within each n-tile ----
  if constexpr (F32OUT) {
    float* C = (float*)Cout;
#pragma unroll
    for (int nt = 0; nt < 16; ++nt) {
#pragma unroll
      for (int r = 0; r < 8; ++r)
        C[(size_t)(m0w + r + 8 * grp) * N + n0b + nt * 16 + row] = acc[nt][r];
    }
  } else {
    _Float16* C = (_Float16*)Cout;
#pragma unroll
    for (int nt = 0; nt < 16; ++nt) {
#pragma unroll
      for (int r = 0; r < 8; ++r)
        C[(size_t)(m0w + r + 8 * grp) * N + n0b + nt * 16 + row] =
            (_Float16)acc[nt][r];
    }
  }
}

// ---------------------------------------------------------------------------
// Flash attention, sliding window (causal, j in [i-WIN, i]), GQA.
// One wave per (batch, head, 16-query tile). Keys processed in blocks of 32.
// S = Q*K^T via 4 chained WMMAs per 16-col tile; P*V via 8 WMMAs over HD.
// ---------------------------------------------------------------------------
__global__ __launch_bounds__(32) void swa_flash_attn(
    const _Float16* __restrict__ Q, const _Float16* __restrict__ Kb,
    const _Float16* __restrict__ Vb, _Float16* __restrict__ Ob) {
  __shared__ __align__(32) _Float16 sQ[16 * 128];
  __shared__ __align__(32) _Float16 sK[32 * 128];   // [key][hd]
  __shared__ __align__(32) _Float16 sVT[128 * 32];  // [hd][key]
  __shared__ __align__(32) _Float16 sP[16 * 32];    // [row][key32]

  const int lane = threadIdx.x & 31;
  const int qt = blockIdx.x & (Ln / 16 - 1);
  const int h = (blockIdx.x >> 7) & (Hn - 1);
  const int b = blockIdx.x >> 11;
  const int kvh = h / (Hn / KVH);
  const int i0 = qt << 4;
  const int row = lane & 15;
  const int grp = lane >> 4;

  // ---- stage Q tile (16 x 128) ----
  {
    const _Float16* qrow =
        Q + ((size_t)(b * Ln + i0 + row) * Hn + h) * HD + grp * 64;
#pragma unroll
    for (int c = 0; c < 4; ++c)
      *(v16h*)&sQ[row * 128 + grp * 64 + c * 16] = *(const v16h*)(qrow + c * 16);
  }
  wait_ds();

  // ---- Q A-fragments for the 4 K-chunks of HD (built once) ----
  v16h qa[4];
#pragma unroll
  for (int c = 0; c < 4; ++c) {
    v8h lo = *(const v8h*)&sQ[row * 128 + c * 32 + grp * 16];
    v8h hi = *(const v8h*)&sQ[row * 128 + c * 32 + grp * 16 + 8];
    qa[c] = __builtin_shufflevector(lo, hi, 0, 1, 2, 3, 4, 5, 6, 7, 8, 9, 10,
                                    11, 12, 13, 14, 15);
  }

  v8f o[8];
#pragma unroll
  for (int hb = 0; hb < 8; ++hb) o[hb] = (v8f){};
  float mrow[8], lrow[8];
#pragma unroll
  for (int r = 0; r < 8; ++r) { mrow[r] = NEGBIG; lrow[r] = 0.f; }

  int jlo = i0 - WIN; if (jlo < 0) jlo = 0;
  const int jb0 = jlo & ~31;

  for (int jb = jb0; jb <= i0 + 15; jb += 32) {
    // ---- stage K (row-major) and V (transposed) tiles, 32 keys ----
    int j = jb + lane; if (j > Ln - 1) j = Ln - 1;  // masked later
    const _Float16* krow = Kb + ((size_t)(b * Ln + j) * KVH + kvh) * HD;
    const _Float16* vrow = Vb + ((size_t)(b * Ln + j) * KVH + kvh) * HD;
#pragma unroll
    for (int c = 0; c < 8; ++c)
      *(v16h*)&sK[lane * 128 + c * 16] = *(const v16h*)(krow + c * 16);
    v16h vv[8];
#pragma unroll
    for (int c = 0; c < 8; ++c) vv[c] = *(const v16h*)(vrow + c * 16);
#pragma unroll
    for (int c = 0; c < 8; ++c) {
#pragma unroll
      for (int e = 0; e < 16; ++e) sVT[(c * 16 + e) * 32 + lane] = vv[c][e];
    }
    if (jb + 32 <= i0 + 15) {
      __builtin_prefetch(krow + (size_t)32 * KVH * HD, 0, 0);
      __builtin_prefetch(vrow + (size_t)32 * KVH * HD, 0, 0);
    }
    wait_ds();

    // ---- S = Q * K^T for two 16-key column tiles ----
    v8f s0 = {}, s1 = {};
#pragma unroll
    for (int c = 0; c < 4; ++c) {
      // B[k][n] = K[key=n][hd=k]: contiguous in sK row-major layout
      v16h b0 = *(const v16h*)&sK[(0 + row) * 128 + c * 32 + grp * 16];
      s0 = wmma_f16(qa[c], b0, s0);
    }
#pragma unroll
    for (int c = 0; c < 4; ++c) {
      v16h b1 = *(const v16h*)&sK[(16 + row) * 128 + c * 32 + grp * 16];
      s1 = wmma_f16(qa[c], b1, s1);
    }

    // ---- scale + sliding-window mask + online softmax ----
    float p0[8], p1[8], alpha[8];
#pragma unroll
    for (int r = 0; r < 8; ++r) {
      const int i = i0 + r + 8 * grp;     // query row of this accumulator slot
      const int j0c = jb + row;           // key col, tile 0
      const int j1c = jb + 16 + row;      // key col, tile 1
      float v0 = s0[r] * SCALE;
      float v1 = s1[r] * SCALE;
      if (j0c > i || j0c < i - WIN) v0 = NEGBIG;
      if (j1c > i || j1c < i - WIN) v1 = NEGBIG;
      float mx = fmaxf(v0, v1);
      mx = fmaxf(mx, __shfl_xor(mx, 1));
      mx = fmaxf(mx, __shfl_xor(mx, 2));
      mx = fmaxf(mx, __shfl_xor(mx, 4));
      mx = fmaxf(mx, __shfl_xor(mx, 8));
      const float mnew = fmaxf(mrow[r], mx);
      const float a = __expf(mrow[r] - mnew);
      const float e0 = (v0 > 0.5f * NEGBIG) ? __expf(v0 - mnew) : 0.f;
      const float e1 = (v1 > 0.5f * NEGBIG) ? __expf(v1 - mnew) : 0.f;
      float sum = e0 + e1;
      sum += __shfl_xor(sum, 1);
      sum += __shfl_xor(sum, 2);
      sum += __shfl_xor(sum, 4);
      sum += __shfl_xor(sum, 8);
      lrow[r] = lrow[r] * a + sum;
      mrow[r] = mnew;
      alpha[r] = a;
      p0[r] = e0;
      p1[r] = e1;
    }

    // ---- rescale running O by alpha (per row) ----
#pragma unroll
    for (int hb = 0; hb < 8; ++hb) {
#pragma unroll
      for (int r = 0; r < 8; ++r) o[hb][r] *= alpha[r];
    }

    // ---- P: C-layout -> LDS -> A-layout fragment ----
#pragma unroll
    for (int r = 0; r < 8; ++r) {
      sP[(r + 8 * grp) * 32 + row] = (_Float16)p0[r];
      sP[(r + 8 * grp) * 32 + 16 + row] = (_Float16)p1[r];
    }
    wait_ds();
    v8h plo = *(const v8h*)&sP[row * 32 + grp * 16];
    v8h phi = *(const v8h*)&sP[row * 32 + grp * 16 + 8];
    v16h pa = __builtin_shufflevector(plo, phi, 0, 1, 2, 3, 4, 5, 6, 7, 8, 9,
                                      10, 11, 12, 13, 14, 15);

    // ---- O += P(16x32) * V(32x128): 8 WMMAs across HD ----
#pragma unroll
    for (int hb = 0; hb < 8; ++hb) {
      // B[k][n] = V[key=k][hd=16*hb+n] = sVT[(16*hb+n)*32 + k], k contiguous
      v16h vb = *(const v16h*)&sVT[(hb * 16 + row) * 32 + grp * 16];
      o[hb] = wmma_f16(pa, vb, o[hb]);
    }
  }

  // ---- epilogue: normalize and store f16 attn output [B, L, H, HD] ----
#pragma unroll
  for (int hb = 0; hb < 8; ++hb) {
#pragma unroll
    for (int r = 0; r < 8; ++r) {
      const float val = o[hb][r] / lrow[r];
      Ob[((size_t)(b * Ln + i0 + r + 8 * grp) * Hn + h) * HD + hb * 16 + row] =
          (_Float16)val;
    }
  }
}

// ---------------------------------------------------------------------------
// Host-side launch
// ---------------------------------------------------------------------------
extern "C" void kernel_launch(void* const* d_in, const int* in_sizes, int n_in,
                              void* d_out, int out_size, void* d_ws,
                              size_t ws_size, hipStream_t stream) {
  (void)in_sizes; (void)n_in; (void)out_size; (void)ws_size;

  const float* x  = (const float*)d_in[0];
  const float* Wq = (const float*)d_in[1];
  const float* Wk = (const float*)d_in[2];
  const float* Wv = (const float*)d_in[3];
  const float* Wo = (const float*)d_in[4];
  float* out = (float*)d_out;

  // Workspace carve-up (~80 MB of f16 staging, 256B aligned regions)
  char* ws = (char*)d_ws;
  size_t off = 0;
  auto carve = [&](size_t halves) {
    _Float16* p = (_Float16*)(ws + off);
    off = (off + halves * sizeof(_Float16) + 255) & ~(size_t)255;
    return p;
  };
  _Float16* hx  = carve((size_t)Bn * Ln * En);
  _Float16* hWq = carve((size_t)En * Hn * HD);
  _Float16* hWk = carve((size_t)En * KVH * HD);
  _Float16* hWv = carve((size_t)En * KVH * HD);
  _Float16* hWo = carve((size_t)Hn * HD * En);
  _Float16* hq  = carve((size_t)Bn * Ln * Hn * HD);
  _Float16* hk  = carve((size_t)Bn * Ln * KVH * HD);
  _Float16* hv  = carve((size_t)Bn * Ln * KVH * HD);
  _Float16* ha  = carve((size_t)Bn * Ln * Hn * HD);

  auto cvt = [&](const float* s, _Float16* d, int n) {
    swa_cvt_f32_to_f16<<<2048, 256, 0, stream>>>(s, d, n);
  };
  cvt(x,  hx,  Bn * Ln * En);
  cvt(Wq, hWq, En * Hn * HD);
  cvt(Wk, hWk, En * KVH * HD);
  cvt(Wv, hWv, En * KVH * HD);
  cvt(Wo, hWo, Hn * HD * En);

  const int M = Bn * Ln;  // batches fuse along M (x is [B,L,E] contiguous)
  // one block (8 waves) per 128x256 C tile
  auto gemm_grid = [](int m, int n) { return (m / 128) * (n / 256); };

  // QKV projections (f16 out)
  swa_gemm_wmma<false><<<gemm_grid(M, Hn * HD), 256, 0, stream>>>(
      hx, hWq, hq, M, Hn * HD, En);
  swa_gemm_wmma<false><<<gemm_grid(M, KVH * HD), 256, 0, stream>>>(
      hx, hWk, hk, M, KVH * HD, En);
  swa_gemm_wmma<false><<<gemm_grid(M, KVH * HD), 256, 0, stream>>>(
      hx, hWv, hv, M, KVH * HD, En);

  // Flash attention: one wave per (b, h, 16-query tile)
  swa_flash_attn<<<Bn * Hn * (Ln / 16), 32, 0, stream>>>(hq, hk, hv, ha);

  // Output projection (f32 out straight to d_out)
  swa_gemm_wmma<true><<<gemm_grid(M, En), 256, 0, stream>>>(
      ha, hWo, (void*)out, M, En, Hn * HD);
}
